// Qwen3MoeSparseMoeBlock_90117003804875
// MI455X (gfx1250) — compile-verified, matrix operand order checked by reading, not run on hardware
//
#include <hip/hip_runtime.h>
#include <hip/hip_bf16.h>

#define NUM_EXPERTS 32
#define TOP_K       4
#define HIDDEN      2048
#define INTER       768
#define N_TOKENS    2048

typedef __attribute__((ext_vector_type(16))) __bf16 v16bf;
typedef __attribute__((ext_vector_type(8)))  float  v8f;

// ---------------------------------------------------------------------------
// Kernel 1: zero output accumulator + per-expert counters
// ---------------------------------------------------------------------------
__global__ void zero_kernel(float* __restrict__ out, size_t n, int* __restrict__ cnt) {
    size_t stride = (size_t)gridDim.x * blockDim.x;
    size_t i = (size_t)blockIdx.x * blockDim.x + threadIdx.x;
    if (i < NUM_EXPERTS) cnt[i] = 0;
    for (; i < n; i += stride) out[i] = 0.0f;
}

// ---------------------------------------------------------------------------
// Kernel 2: router. One wave32 per token; lane e owns expert e.
// ---------------------------------------------------------------------------
__global__ void router_kernel(const float* __restrict__ hs,
                              const float* __restrict__ wg,
                              int*   __restrict__ cnt,
                              int*   __restrict__ toklist,
                              float* __restrict__ wlist) {
    const int token = blockIdx.x;
    const int lane  = threadIdx.x;              // 0..31 == expert id

    const float* __restrict__ x = hs + (size_t)token * HIDDEN;
    float acc = 0.0f;
    for (int k = 0; k < HIDDEN; k += 4) {
        float4 xv = *(const float4*)(x + k);
        acc += xv.x * wg[(k + 0) * NUM_EXPERTS + lane];
        acc += xv.y * wg[(k + 1) * NUM_EXPERTS + lane];
        acc += xv.z * wg[(k + 2) * NUM_EXPERTS + lane];
        acc += xv.w * wg[(k + 3) * NUM_EXPERTS + lane];
    }

    // softmax numerator (denominator cancels in top-k renorm)
    float vmax = acc;
    #pragma unroll
    for (int off = 16; off >= 1; off >>= 1)
        vmax = fmaxf(vmax, __shfl_xor(vmax, off, 32));
    float pv = __expf(acc - vmax);

    float topV[TOP_K];
    int   topI[TOP_K];
    float sum4 = 0.0f;
    #pragma unroll
    for (int i = 0; i < TOP_K; ++i) {
        float v = pv; int ix = lane;
        #pragma unroll
        for (int off = 16; off >= 1; off >>= 1) {
            float ov = __shfl_xor(v,  off, 32);
            int   oi = __shfl_xor(ix, off, 32);
            if (ov > v || (ov == v && oi < ix)) { v = ov; ix = oi; }
        }
        topV[i] = v; topI[i] = ix; sum4 += v;
        if (lane == ix) pv = -1.0f;
    }

    if (lane == 0) {
        float rs = __builtin_amdgcn_rcpf(sum4);
        #pragma unroll
        for (int i = 0; i < TOP_K; ++i) {
            int e   = topI[i];
            int pos = atomicAdd(&cnt[e], 1);
            toklist[e * N_TOKENS + pos] = token;
            wlist  [e * N_TOKENS + pos] = topV[i] * rs;
        }
    }
}

// ---------------------------------------------------------------------------
// Kernel 3: fused expert: gate_up GEMM -> SiLU*up*w -> down GEMM (atomic add)
// grid = (tokenTiles=128, experts=32), block = 256 (8 wave32s)
// ---------------------------------------------------------------------------
__device__ __forceinline__ void cvt16(v16bf& d, float4 a0, float4 a1,
                                      float4 a2, float4 a3) {
    d[0]=(__bf16)a0.x; d[1]=(__bf16)a0.y; d[2]=(__bf16)a0.z; d[3]=(__bf16)a0.w;
    d[4]=(__bf16)a1.x; d[5]=(__bf16)a1.y; d[6]=(__bf16)a1.z; d[7]=(__bf16)a1.w;
    d[8]=(__bf16)a2.x; d[9]=(__bf16)a2.y; d[10]=(__bf16)a2.z; d[11]=(__bf16)a2.w;
    d[12]=(__bf16)a3.x; d[13]=(__bf16)a3.y; d[14]=(__bf16)a3.z; d[15]=(__bf16)a3.w;
}

__global__ __launch_bounds__(256, 1)
void moe_expert_kernel(const float* __restrict__ hs,
                       const float* __restrict__ Wgu,
                       const float* __restrict__ Wd,
                       const int*   __restrict__ cnt,
                       const int*   __restrict__ toklist,
                       const float* __restrict__ wlist,
                       float*       __restrict__ out) {
    // Row pad of 4 floats (16B) -> consecutive M rows land on different banks.
    __shared__ float xs[16][HIDDEN + 4];   // 16-token fp32 activation tile (~128 KB)
    __shared__ float ys[16][INTER + 4];    // SiLU(g)*u*w intermediate (~48 KB)
    __shared__ int   tokS[16];
    __shared__ float wtS[16];

    const int e    = blockIdx.y;
    const int m0   = blockIdx.x * 16;
    const int tcnt = cnt[e];
    if (m0 >= tcnt) return;                         // block-uniform exit

    const int tid = threadIdx.x;
    if (tid < 16) {
        int idx = m0 + tid;
        if (idx < tcnt) { tokS[tid] = toklist[e * N_TOKENS + idx];
                          wtS[tid]  = wlist  [e * N_TOKENS + idx]; }
        else            { tokS[tid] = 0; wtS[tid] = 0.0f; }
    }
    __syncthreads();

    // Cooperative stage of 16 x HIDDEN fp32 activations into LDS.
    for (int v = tid; v < 16 * (HIDDEN / 4); v += 256) {
        int row = v >> 9;                 // /512
        int c4  = v & 511;
        float4 val = *(const float4*)(hs + (size_t)tokS[row] * HIDDEN + c4 * 4);
        *(float4*)&xs[row][c4 * 4] = val;
    }
    __syncthreads();

    const int wv   = tid >> 5;
    const int lane = tid & 31;
    const int lm   = lane & 15;
    const bool hi  = lane >= 16;
    const int koff = hi ? 8 : 0;          // ISA 16-bit A/B lane-half K split
    const int mC   = hi ? 8 : 0;          // C/D: VGPR i -> M = i + mC

    // ---------------- Phase 1: gate_up + SiLU ----------------
    const size_t gustride = 2 * INTER;    // row stride of Wgu[e] in floats
    const float* WguE = Wgu + (size_t)e * HIDDEN * gustride;

    for (int p = wv; p < INTER / 16; p += 8) {       // 48 column tiles
        const int cg = p * 16;
        v8f accg = {}; v8f accu = {};
        const float* bptr = WguE + (size_t)koff * gustride + cg + lm;

        for (int k0 = 0; k0 < HIDDEN; k0 += 32) {
            // Prefetch the weight stream two K-blocks ahead (global_prefetch_b8;
            // speculative, OOB translation failures are silently dropped).
            __builtin_prefetch(bptr + 64 * gustride, 0, 1);
            __builtin_prefetch(bptr + 64 * gustride + INTER, 0, 1);

            // A fragment from LDS (fp32 -> bf16)
            const float* xr = &xs[lm][k0 + koff];
            v16bf a;
            cvt16(a, *(const float4*)(xr),      *(const float4*)(xr + 4),
                     *(const float4*)(xr + 16), *(const float4*)(xr + 20));
            // B fragments (gate + up), strided fp32 loads -> bf16
            v16bf bg, bu;
            const float* bp = bptr;
            #pragma unroll
            for (int j = 0; j < 8; ++j) {
                bg[j]     = (__bf16)bp[0];
                bu[j]     = (__bf16)bp[INTER];
                bg[8 + j] = (__bf16)bp[16 * gustride];
                bu[8 + j] = (__bf16)bp[16 * gustride + INTER];
                bp += gustride;
            }
            bptr += 32 * gustride;

            accg = __builtin_amdgcn_wmma_f32_16x16x32_bf16(
                       false, a, false, bg, (short)0, accg, false, false);
            accu = __builtin_amdgcn_wmma_f32_16x16x32_bf16(
                       false, a, false, bu, (short)0, accu, false, false);
        }
        #pragma unroll
        for (int i = 0; i < 8; ++i) {
            int m = i + mC;
            float g = accg[i], u = accu[i];
            // SiLU via hardware v_rcp_f32 (avoids IEEE div expansion)
            float sig = __builtin_amdgcn_rcpf(1.0f + __expf(-g));
            ys[m][cg + lm] = g * sig * u * wtS[m];
        }
    }
    __syncthreads();

    // ---------------- Phase 2: down projection ----------------
    const float* WdE = Wd + (size_t)e * INTER * HIDDEN;

    for (int t = wv; t < HIDDEN / 16; t += 8) {      // 128 column tiles
        const int cn = t * 16;
        v8f acc = {};
        const float* bptr = WdE + (size_t)koff * HIDDEN + cn + lm;

        for (int k0 = 0; k0 < INTER; k0 += 32) {
            __builtin_prefetch(bptr + 64 * HIDDEN, 0, 1);

            const float* yr = &ys[lm][k0 + koff];
            v16bf a;
            cvt16(a, *(const float4*)(yr),      *(const float4*)(yr + 4),
                     *(const float4*)(yr + 16), *(const float4*)(yr + 20));
            v16bf b;
            const float* bp = bptr;
            #pragma unroll
            for (int j = 0; j < 8; ++j) {
                b[j]     = (__bf16)bp[0];
                b[8 + j] = (__bf16)bp[16 * HIDDEN];
                bp += HIDDEN;
            }
            bptr += 32 * HIDDEN;

            acc = __builtin_amdgcn_wmma_f32_16x16x32_bf16(
                      false, a, false, b, (short)0, acc, false, false);
        }
        #pragma unroll
        for (int i = 0; i < 8; ++i) {
            int m = i + mC;
            unsafeAtomicAdd(&out[(size_t)tokS[m] * HIDDEN + cn + lm], acc[i]);
        }
    }
}

// ---------------------------------------------------------------------------
extern "C" void kernel_launch(void* const* d_in, const int* in_sizes, int n_in,
                              void* d_out, int out_size, void* d_ws, size_t ws_size,
                              hipStream_t stream) {
    const float* hs  = (const float*)d_in[0];   // (N, H)
    const float* Wg  = (const float*)d_in[1];   // (H, E)
    const float* Wgu = (const float*)d_in[2];   // (E, H, 2I)
    const float* Wd  = (const float*)d_in[3];   // (E, I, H)
    float* out = (float*)d_out;                 // (N, H) fp32

    // workspace layout
    int*   cnt     = (int*)d_ws;
    int*   toklist = cnt + NUM_EXPERTS;
    float* wlist   = (float*)(toklist + NUM_EXPERTS * N_TOKENS);

    size_t outElems = (size_t)N_TOKENS * HIDDEN;
    zero_kernel<<<4096, 256, 0, stream>>>(out, outElems, cnt);
    router_kernel<<<N_TOKENS, 32, 0, stream>>>(hs, Wg, cnt, toklist, wlist);

    dim3 grid(N_TOKENS / 16, NUM_EXPERTS);
    moe_expert_kernel<<<grid, 256, 0, stream>>>(hs, Wgu, Wd, cnt, toklist, wlist, out);
}
